// ViewTexUNet_44994077393504
// MI455X (gfx1250) — matrix operand order repaired and checked by reading, not compiled, not run on hardware
//
#include <hip/hip_runtime.h>

typedef __attribute__((ext_vector_type(16))) _Float16 v16h;
typedef __attribute__((ext_vector_type(8)))  _Float16 v8h;
typedef __attribute__((ext_vector_type(8)))  float    v8f;

// ---------------------------------------------------------------------------
// small vector helpers
// ---------------------------------------------------------------------------
struct V3 { float x, y, z; };
__device__ __forceinline__ V3 mk3(float a, float b, float c) { return {a, b, c}; }
__device__ __forceinline__ V3 sub3(V3 a, V3 b) { return {a.x - b.x, a.y - b.y, a.z - b.z}; }
__device__ __forceinline__ V3 cross3(V3 a, V3 b) {
  return {a.y * b.z - a.z * b.y, a.z * b.x - a.x * b.z, a.x * b.y - a.y * b.x};
}
__device__ __forceinline__ float dot3(V3 a, V3 b) { return a.x * b.x + a.y * b.y + a.z * b.z; }
__device__ __forceinline__ V3 norm3(V3 a) {
  float n = sqrtf(dot3(a, a));
  float d = fmaxf(n, 1e-8f);
  return {a.x / d, a.y / d, a.z / d};
}

// ---------------------------------------------------------------------------
// weight prep: f32 (co,ciSrc,4,4) -> f16 [coPad][K = ciPad*16]
// K ordering: k = (kh*4+kw)*ciPad + ci   (pos-major, ci-minor)
// ---------------------------------------------------------------------------
__global__ void wprep(const float* __restrict__ src, _Float16* __restrict__ dst,
                      int Co, int CiSrc, int CiPad, int CoPad, int lcp) {
  int t = blockIdx.x * blockDim.x + threadIdx.x;
  int Kp = CiPad << 4;
  int total = CoPad * Kp;
  if (t >= total) return;
  int n = t >> (lcp + 4);
  int k = t & (Kp - 1);
  int pos = k >> lcp;
  int ci = k & (CiPad - 1);
  float v = 0.f;
  if (n < Co && ci < CiSrc) v = src[((size_t)n * CiSrc + ci) * 16 + pos];
  dst[t] = (_Float16)v;
}

// ---------------------------------------------------------------------------
// geometry: TBN + view blend -> cond, NHWC with 8 channels (6 real + 2 zero)
// ---------------------------------------------------------------------------
__global__ void geom_kernel(const float* __restrict__ verts, const float* __restrict__ tex,
                            const float* __restrict__ cam, const int* __restrict__ idximg,
                            const float* __restrict__ uvc, const int* __restrict__ uvmap,
                            const float* __restrict__ bary, _Float16* __restrict__ cond,
                            int Bn, int Nv, int Hn) {
  int pix = blockIdx.x * blockDim.x + threadIdx.x;
  int total = Hn * Hn;
  if (pix >= total) return;

  int i0 = idximg[3 * pix + 0];
  int i1 = idximg[3 * pix + 1];
  int i2 = idximg[3 * pix + 2];
  float maskf = (i0 != -1 || i1 != -1 || i2 != -1) ? 1.f : 0.f;
  int j0 = i0 < 0 ? 0 : i0;
  int j1 = i1 < 0 ? 0 : i1;
  int j2 = i2 < 0 ? 0 : i2;

  int u0i = uvmap[2 * j0], u1i = uvmap[2 * j1], u2i = uvmap[2 * j2];
  float u0x = uvc[2 * u0i], u0y = uvc[2 * u0i + 1];
  float u1x = uvc[2 * u1i], u1y = uvc[2 * u1i + 1];
  float u2x = uvc[2 * u2i], u2y = uvc[2 * u2i + 1];
  float vt01y = u1y - u0y;
  float vt02y = u2y - u0y;
  float den = (u1x - u0x) * vt02y - vt01y * (u2x - u0x);
  float f = 1.f / (den + (den >= 0.f ? 1e-8f : -1e-8f));

  float b0 = bary[3 * pix + 0], b1 = bary[3 * pix + 1], b2 = bary[3 * pix + 2];

  for (int b = 0; b < Bn; ++b) {
    const float* vb = verts + (size_t)b * Nv * 3;
    V3 p0 = mk3(vb[3 * j0], vb[3 * j0 + 1], vb[3 * j0 + 2]);
    V3 p1 = mk3(vb[3 * j1], vb[3 * j1 + 1], vb[3 * j1 + 2]);
    V3 p2 = mk3(vb[3 * j2], vb[3 * j2 + 1], vb[3 * j2 + 2]);
    V3 v01 = sub3(p1, p0), v02 = sub3(p2, p0);
    V3 n = norm3(cross3(v01, v02));
    V3 t = norm3(mk3(f * (vt02y * v01.x - vt01y * v02.x),
                     f * (vt02y * v01.y - vt01y * v02.y),
                     f * (vt02y * v01.z - vt01y * v02.z)));
    V3 bb = norm3(cross3(n, t));
    V3 cp = mk3(cam[3 * b + 0], cam[3 * b + 1], cam[3 * b + 2]);
    V3 w0 = norm3(sub3(cp, p0));
    V3 w1 = norm3(sub3(cp, p1));
    V3 w2 = norm3(sub3(cp, p2));
    V3 vu = mk3(w0.x * b0 + w1.x * b1 + w2.x * b2,
                w0.y * b0 + w1.y * b1 + w2.y * b2,
                w0.z * b0 + w1.z * b1 + w2.z * b2);
    const float* tb = tex + (size_t)b * 3 * total + pix;
    v8h cv;
    cv[0] = (_Float16)(dot3(t, vu) * maskf);
    cv[1] = (_Float16)(-dot3(bb, vu) * maskf);
    cv[2] = (_Float16)(dot3(n, vu) * maskf);
    cv[3] = (_Float16)tb[0 * (size_t)total];
    cv[4] = (_Float16)tb[1 * (size_t)total];
    cv[5] = (_Float16)tb[2 * (size_t)total];
    cv[6] = (_Float16)0.f;
    cv[7] = (_Float16)0.f;
    *(v8h*)(cond + ((size_t)b * total + pix) * 8) = cv;
  }
}

// ---------------------------------------------------------------------------
// per-lane 8-half contiguous A-fragment runs (NHWC activations)
// ---------------------------------------------------------------------------
__device__ __forceinline__ v8h load_run_down(const _Float16* __restrict__ xb, int ks, int lci,
                                             int oh, int owp, int Hin) {
  int pos = ks >> lci;
  int cis = ks & ((1 << lci) - 1);
  int ih = 2 * oh - 1 + (pos >> 2);
  int iw = 2 * owp - 1 + (pos & 3);
  if ((unsigned)ih < (unsigned)Hin && (unsigned)iw < (unsigned)Hin)
    return *(const v8h*)(xb + ((((size_t)ih * Hin + iw) << lci) + cis));
  v8h z = {};
  return z;
}

__device__ __forceinline__ v8h load_run_up(const _Float16* __restrict__ xb, int ks, int lci,
                                           int oh, int owp, int Hin) {
  int pos = ks >> lci;
  int cis = ks & ((1 << lci) - 1);
  int ih2 = oh - 2 + (pos >> 2);
  int iw2 = owp - 2 + (pos & 3);
  if (ih2 >= 0 && iw2 >= 0 && !((ih2 | iw2) & 1)) {
    int ih = ih2 >> 1, iw = iw2 >> 1;
    if (ih < Hin && iw < Hin)
      return *(const v8h*)(xb + ((((size_t)ih * Hin + iw) << lci) + cis));
  }
  v8h z = {};
  return z;
}

// stage this block's 16 x K f16 weight tile into LDS with async DMA copies
__device__ __forceinline__ void stage_weights(const _Float16* __restrict__ wt,
                                              _Float16* ldsw, int K) {
  unsigned ldsBase = (unsigned)(unsigned long long)(void*)ldsw;
  int n128 = K * 2;  // 16*K halves = 32K bytes = 2K x 16B chunks
  for (int idx = (int)threadIdx.x; idx < n128; idx += 256) {
    unsigned lo = ldsBase + (unsigned)idx * 16u;
    unsigned long long ga = (unsigned long long)(const void*)((const char*)wt + (size_t)idx * 16);
    asm volatile("global_load_async_to_lds_b128 %0, %1, off" :: "v"(lo), "v"(ga) : "memory");
  }
  asm volatile("s_wait_asynccnt 0x0" ::: "memory");
  __syncthreads();
}

// ---------------------------------------------------------------------------
// down conv 4x4 stride2 pad1 as implicit GEMM (NHWC, f16 in, f32 acc)
// grid: (pixel-tile tasks/8, CoPad/16, B); block 256 = 8 waves, 1 tile/wave
// ---------------------------------------------------------------------------
__global__ void down_conv_wmma(const _Float16* __restrict__ x, const _Float16* __restrict__ wT,
                               const float* __restrict__ bias, _Float16* __restrict__ out,
                               int lci, int Co, int Hin, int Hout, int ltw, int tasks) {
  extern __shared__ _Float16 ldsw[];
  int K = 16 << lci;
  stage_weights(wT + ((size_t)blockIdx.y << 4) * K, ldsw, K);

  int task = blockIdx.x * 8 + (threadIdx.x >> 5);
  if (task >= tasks) return;                 // wave-uniform: EXEC stays all-1s
  int lane = threadIdx.x & 31;
  int grp = lane >> 4;
  int l16 = lane & 15;
  int tx = task & ((1 << ltw) - 1);
  int oh = task >> ltw;
  int ow0 = tx << 4;
  int owp = ow0 + l16;

  const _Float16* xb = x + (((size_t)blockIdx.z * Hin * Hin) << lci);
  const _Float16* lw = ldsw + (size_t)l16 * K + grp * 8;

  v8f acc = {};
  for (int k0 = 0; k0 < K; k0 += 32) {
    int ks0 = k0 + (grp << 3);
    v8h r0 = load_run_down(xb, ks0, lci, oh, owp, Hin);
    v8h r1 = load_run_down(xb, ks0 + 16, lci, oh, owp, Hin);
    v8h b0 = *(const v8h*)(lw + k0);
    v8h b1 = *(const v8h*)(lw + k0 + 16);
    v16h af, bf;
#pragma unroll
    for (int i = 0; i < 8; ++i) { af[i] = r0[i]; af[8 + i] = r1[i]; bf[i] = b0[i]; bf[8 + i] = b1[i]; }
    acc = __builtin_amdgcn_wmma_f32_16x16x32_f16(false, af, false, bf, (short)0, acc,
                                                 false, false);
  }

  int n = ((int)blockIdx.y << 4) + l16;
  if (n < Co) {
#pragma unroll
    for (int v = 0; v < 8; ++v) {
      int ow_ = ow0 + grp * 8 + v;
      float val = acc[v] + bias[((size_t)n * Hout + oh) * Hout + ow_];
      val = val > 0.f ? val : 0.2f * val;
      out[(((size_t)blockIdx.z * Hout * Hout) + (size_t)oh * Hout + ow_) * Co + n] = (_Float16)val;
    }
  }
}

// ---------------------------------------------------------------------------
// up (transposed) conv: lhs_dilation=2, 4x4, pad2 + skip add (NHWC)
// ---------------------------------------------------------------------------
__global__ void up_conv_wmma(const _Float16* __restrict__ x, const _Float16* __restrict__ wT,
                             const float* __restrict__ bias, const _Float16* __restrict__ skip,
                             _Float16* __restrict__ out,
                             int lci, int Co, int Hin, int Hout, int ltw, int tasks) {
  extern __shared__ _Float16 ldsw[];
  int K = 16 << lci;
  stage_weights(wT + ((size_t)blockIdx.y << 4) * K, ldsw, K);

  int task = blockIdx.x * 8 + (threadIdx.x >> 5);
  if (task >= tasks) return;
  int lane = threadIdx.x & 31;
  int grp = lane >> 4;
  int l16 = lane & 15;
  int tx = task & ((1 << ltw) - 1);
  int oh = task >> ltw;
  int ow0 = tx << 4;
  int owp = ow0 + l16;

  const _Float16* xb = x + (((size_t)blockIdx.z * Hin * Hin) << lci);
  const _Float16* lw = ldsw + (size_t)l16 * K + grp * 8;

  v8f acc = {};
  for (int k0 = 0; k0 < K; k0 += 32) {
    int ks0 = k0 + (grp << 3);
    v8h r0 = load_run_up(xb, ks0, lci, oh, owp, Hin);
    v8h r1 = load_run_up(xb, ks0 + 16, lci, oh, owp, Hin);
    v8h b0 = *(const v8h*)(lw + k0);
    v8h b1 = *(const v8h*)(lw + k0 + 16);
    v16h af, bf;
#pragma unroll
    for (int i = 0; i < 8; ++i) { af[i] = r0[i]; af[8 + i] = r1[i]; bf[i] = b0[i]; bf[8 + i] = b1[i]; }
    acc = __builtin_amdgcn_wmma_f32_16x16x32_f16(false, af, false, bf, (short)0, acc,
                                                 false, false);
  }

  int n = ((int)blockIdx.y << 4) + l16;
  if (n < Co) {
#pragma unroll
    for (int v = 0; v < 8; ++v) {
      int ow_ = ow0 + grp * 8 + v;
      size_t oidx = (((size_t)blockIdx.z * Hout * Hout) + (size_t)oh * Hout + ow_) * Co + n;
      float val = acc[v] + bias[((size_t)n * Hout + oh) * Hout + ow_];
      val = val > 0.f ? val : 0.2f * val;
      if (skip) val += (float)skip[oidx];
      out[oidx] = (_Float16)val;
    }
  }
}

// ---------------------------------------------------------------------------
// final 1x1 conv (8 -> 3) + 0.1*(.+ob); NHWC f16 in, NCHW f32 out
// ---------------------------------------------------------------------------
__global__ void final_conv(const _Float16* __restrict__ y, const float* __restrict__ oww,
                           const float* __restrict__ ob, float* __restrict__ out,
                           int Hn, int Bn) {
  int pix = blockIdx.x * blockDim.x + threadIdx.x;
  int total = Hn * Hn;
  if (pix >= Bn * total) return;
  int b = pix / total;
  int p = pix - b * total;
  v8h fv = *(const v8h*)(y + (size_t)pix * 8);
#pragma unroll
  for (int c = 0; c < 3; ++c) {
    float acc = 0.f;
#pragma unroll
    for (int q = 0; q < 8; ++q) acc += (float)fv[q] * oww[c * 8 + q];
    out[((size_t)b * 3 + c) * total + p] = 0.1f * (acc + ob[(size_t)c * total + p]);
  }
}

// ---------------------------------------------------------------------------
// host side
// ---------------------------------------------------------------------------
extern "C" void kernel_launch(void* const* d_in, const int* in_sizes, int n_in,
                              void* d_out, int out_size, void* d_ws, size_t ws_size,
                              hipStream_t stream) {
  (void)in_sizes; (void)n_in; (void)out_size; (void)ws_size;
  const float* verts = (const float*)d_in[0];
  const float* tex   = (const float*)d_in[1];
  const float* cam   = (const float*)d_in[2];
  const int*   idxim = (const int*)d_in[3];
  const float* uvc   = (const float*)d_in[4];
  const int*   uvmap = (const int*)d_in[5];
  const float* bary  = (const float*)d_in[6];
  const float *dw[5], *db[5], *uw[5], *ub[5];
  for (int i = 0; i < 5; ++i) {
    dw[i] = (const float*)d_in[7 + 2 * i];
    db[i] = (const float*)d_in[8 + 2 * i];
    uw[i] = (const float*)d_in[17 + 2 * i];
    ub[i] = (const float*)d_in[18 + 2 * i];
  }
  const float* oww = (const float*)d_in[27];
  const float* obb = (const float*)d_in[28];
  float* outp = (float*)d_out;

  const int Bn = 2, Nv = 30000, Hn = 1024;
  const int dCiS[5] = {6, 8, 16, 32, 64},  dCiP[5] = {8, 8, 16, 32, 64},  dCo[5] = {8, 16, 32, 64, 128};
  const int uCiP[5] = {128, 64, 32, 16, 8}, uCo[5] = {64, 32, 16, 8, 8};

  _Float16* wsb = (_Float16*)d_ws;
  size_t off = 0;
  auto alloc = [&](size_t n) { size_t c = off; off += (n + 127) & ~(size_t)127; return c; };

  size_t dwsO[5], uwsO[5];
  int dCoP[5], uCoP[5];
  for (int i = 0; i < 5; ++i) { dCoP[i] = (dCo[i] + 15) & ~15; dwsO[i] = alloc((size_t)dCoP[i] * dCiP[i] * 16); }
  for (int i = 0; i < 5; ++i) { uCoP[i] = (uCo[i] + 15) & ~15; uwsO[i] = alloc((size_t)uCoP[i] * uCiP[i] * 16); }

  size_t condO = alloc((size_t)Bn * 8 * Hn * Hn);          // NHWC, ci padded to 8
  size_t x2o = alloc((size_t)Bn * 8 * 512 * 512);
  size_t x3o = alloc((size_t)Bn * 16 * 256 * 256);
  size_t x4o = alloc((size_t)Bn * 32 * 128 * 128);
  size_t x5o = alloc((size_t)Bn * 64 * 64 * 64);
  size_t x6o = alloc((size_t)Bn * 128 * 32 * 32);
  size_t y1o = alloc((size_t)Bn * 8 * 1024 * 1024);
  // decoder intermediates alias the (dead after down0) cond region
  size_t y5o = condO;
  size_t y4o = y5o + (size_t)Bn * 64 * 64 * 64;
  size_t y3o = y4o + (size_t)Bn * 32 * 128 * 128;
  size_t y2o = y3o + (size_t)Bn * 16 * 256 * 256;

  // ---- weight conversion (f32 OIHW -> f16 [coPad][pos-major, ci-minor K])
  for (int i = 0; i < 5; ++i) {
    int Kp = dCiP[i] * 16, tot = dCoP[i] * Kp;
    wprep<<<(tot + 255) / 256, 256, 0, stream>>>(dw[i], wsb + dwsO[i], dCo[i], dCiS[i], dCiP[i],
                                                 dCoP[i], __builtin_ctz(dCiP[i]));
  }
  for (int i = 0; i < 5; ++i) {
    int Kp = uCiP[i] * 16, tot = uCoP[i] * Kp;
    wprep<<<(tot + 255) / 256, 256, 0, stream>>>(uw[i], wsb + uwsO[i], uCo[i], uCiP[i], uCiP[i],
                                                 uCoP[i], __builtin_ctz(uCiP[i]));
  }

  // ---- geometry -> cond
  int totalPix = Hn * Hn;
  geom_kernel<<<(totalPix + 255) / 256, 256, 0, stream>>>(verts, tex, cam, idxim, uvc, uvmap,
                                                          bary, wsb + condO, Bn, Nv, Hn);

  // ---- encoder
  auto down = [&](size_t xo, int li, size_t oo, int Hin) {
    int Hout = Hin / 2, tw = Hout >> 4;
    int tasks = tw * Hout;
    int K = dCiP[li] * 16;
    dim3 g((tasks + 7) / 8, dCoP[li] / 16, Bn);
    down_conv_wmma<<<g, 256, (size_t)K * 32, stream>>>(wsb + xo, wsb + dwsO[li], db[li], wsb + oo,
                                                       __builtin_ctz(dCiP[li]), dCo[li], Hin, Hout,
                                                       __builtin_ctz(tw), tasks);
  };
  down(condO, 0, x2o, 1024);
  down(x2o, 1, x3o, 512);
  down(x3o, 2, x4o, 256);
  down(x4o, 3, x5o, 128);
  down(x5o, 4, x6o, 64);

  // ---- decoder
  auto up = [&](size_t xo, int li, size_t so, bool hasSkip, size_t oo, int Hin) {
    int Hout = Hin * 2, tw = Hout >> 4;
    int tasks = tw * Hout;
    int K = uCiP[li] * 16;
    dim3 g((tasks + 7) / 8, uCoP[li] / 16, Bn);
    up_conv_wmma<<<g, 256, (size_t)K * 32, stream>>>(
        wsb + xo, wsb + uwsO[li], ub[li],
        hasSkip ? (const _Float16*)(wsb + so) : (const _Float16*)nullptr,
        wsb + oo, __builtin_ctz(uCiP[li]), uCo[li], Hin, Hout, __builtin_ctz(tw), tasks);
  };
  up(x6o, 0, x5o, true, y5o, 32);
  up(y5o, 1, x4o, true, y4o, 64);
  up(y4o, 2, x3o, true, y3o, 128);
  up(y3o, 3, x2o, true, y2o, 256);
  up(y2o, 4, 0, false, y1o, 512);

  // ---- output head
  final_conv<<<(Bn * Hn * Hn + 255) / 256, 256, 0, stream>>>(wsb + y1o, oww, obb, outp, Hn, Bn);
}
